// SampleKSoftmaxUnbiasedWithTrimmedLassoGate_43602507989093
// MI455X (gfx1250) — matrix operand order repaired and checked by reading, not compiled, not run on hardware
//
#include <hip/hip_runtime.h>
#include <hip/hip_bf16.h>

// ---------------------------------------------------------------------------
// MoE top-2 gate (sample-K softmax, debiased) + expert combine for MI455X.
//
// Perf model: streaming h (402.7 MB) dominates -> memory-bound (~22 us at
// 23.3 TB/s). Gate GEMM [16384x768]x[768x8] is tiny; we run it on the WMMA
// pipe (v_wmma_f32_16x16x32_f16) with f16 inputs / f32 accumulation staged
// through LDS. The combine streams h with non-temporal loads so the 402 MB
// one-shot stream does not thrash the 192 MB L2.
// ---------------------------------------------------------------------------

typedef __attribute__((ext_vector_type(16))) _Float16 v16h;
typedef __attribute__((ext_vector_type(8)))  _Float16 v8h;
typedef __attribute__((ext_vector_type(4)))  _Float16 v4h;
typedef __attribute__((ext_vector_type(8)))  float    v8f;
typedef __attribute__((ext_vector_type(4)))  float    f4;

#define D_DIM 768
#define E_DIM 8
#define TAU   1.0f
#define K_SEL 2
#define EPS_G 1e-10f

// ---------------------------------------------------------------------------
// Kernel 1: gate. One wave (32 threads) per block, 16 rows of x per block.
//   - stage x tile (16x768) and zero-padded W^T tile (16x768) in LDS as f16
//   - 24 x v_wmma_f32_16x16x32_f16 accumulating 16x16 f32 logits
//   - 16 lanes run per-row softmax / top-2 / debiased masked softmax
//   - write sparse gates g' [16 x 8] f32 to workspace
// ---------------------------------------------------------------------------
__global__ __launch_bounds__(32) void moe_gate_wmma_kernel(
    const float* __restrict__ x,        // [B, 768]
    const float* __restrict__ gate_w,   // [8, 768]
    const float* __restrict__ bias,     // [8]
    float* __restrict__ gs)             // [B, 8] sparse gates (out)
{
    __shared__ _Float16 xs [16 * D_DIM];   // 24 KB  x tile (f16)
    __shared__ _Float16 wsm[16 * D_DIM];   // 24 KB  W^T tile, rows 8..15 = 0
    __shared__ float    lt [16 * 16];      // logits tile for epilogue

    const int lane    = threadIdx.x;       // 0..31 (wave32)
    const int rowBase = blockIdx.x * 16;

    // ---- stage x tile: coalesced f32x4 loads, convert to f16, 8B LDS stores
    const f4* xg4 = (const f4*)(x + (size_t)rowBase * D_DIM);
    for (int i = lane; i < (16 * D_DIM) / 4; i += 32) {
        f4 v = xg4[i];
        v4h hv = {(_Float16)v.x, (_Float16)v.y, (_Float16)v.z, (_Float16)v.w};
        *(v4h*)&xs[i * 4] = hv;
    }
    // ---- stage gate weights (8 real rows + 8 zero rows for N-padding)
    const f4* wg4 = (const f4*)gate_w;
    for (int i = lane; i < (E_DIM * D_DIM) / 4; i += 32) {
        f4 v = wg4[i];
        v4h hv = {(_Float16)v.x, (_Float16)v.y, (_Float16)v.z, (_Float16)v.w};
        *(v4h*)&wsm[i * 4] = hv;
    }
    {
        v4h z = {};
        for (int i = lane; i < (E_DIM * D_DIM) / 4; i += 32)
            *(v4h*)&wsm[E_DIM * D_DIM + i * 4] = z;
    }
    __syncthreads();

    // ---- WMMA main loop: D = A(16x32 f16) * B(32x16 f16) + C(16x16 f32)
    const int m   = lane & 15;           // A row / D column index for this lane
    const int n   = lane & 15;           // B column
    const int sel = lane >> 4;           // 0: lanes 0-15, 1: lanes 16-31

    const _Float16* xrow = &xs [m * D_DIM];
    const _Float16* wrow = &wsm[n * D_DIM];

    v8f c = {};
    #pragma unroll
    for (int k0 = 0; k0 < D_DIM; k0 += 32) {
        // A frag (ISA 16-bit A 16x32 layout):
        //   lanes 0-15:  halves 0-7 -> K=k0..k0+7,   halves 8-15 -> K=k0+16..k0+23
        //   lanes 16-31: halves 0-7 -> K=k0+8..+15,  halves 8-15 -> K=k0+24..+31
        v8h alo = *(const v8h*)(xrow + k0 + sel * 8);
        v8h ahi = *(const v8h*)(xrow + k0 + 16 + sel * 8);
        v16h a = __builtin_shufflevector(alo, ahi,
            0,1,2,3,4,5,6,7,8,9,10,11,12,13,14,15);

        // B frag (32x16): same K order as A: lanes 0-15 K=k0..+7, k0+16..+23,
        //                 lanes 16-31 K=k0+8..+15, k0+24..+31
        v8h blo = *(const v8h*)(wrow + k0 + sel * 8);
        v8h bhi = *(const v8h*)(wrow + k0 + 16 + sel * 8);
        v16h b = __builtin_shufflevector(blo, bhi,
            0,1,2,3,4,5,6,7,8,9,10,11,12,13,14,15);

        c = __builtin_amdgcn_wmma_f32_16x16x32_f16(
                /*neg_a=*/false, a, /*neg_b=*/false, b,
                /*c_mod=*/(short)0, c, /*reuse_a=*/false, /*reuse_b=*/false);
    }

    // ---- spill D tile: lane, VGPR r -> row (r + 8*sel), col (lane&15)
    #pragma unroll
    for (int r = 0; r < 8; ++r)
        lt[(r + 8 * sel) * 16 + n] = c[r];
    __syncthreads();

    // ---- per-row gating epilogue: 16 lanes, one row each
    if (lane < 16) {
        const int row = rowBase + lane;
        float lg[E_DIM], p[E_DIM];
        float mx = -3.4e38f;
        #pragma unroll
        for (int e = 0; e < E_DIM; ++e) {
            lg[e] = (lt[lane * 16 + e] + bias[e]) * (1.0f / TAU);
            mx = fmaxf(mx, lg[e]);
        }
        float sum = 0.f;
        #pragma unroll
        for (int e = 0; e < E_DIM; ++e) { p[e] = __expf(lg[e] - mx); sum += p[e]; }
        const float inv_sum = 1.0f / sum;

        // top-2 on logits (ties -> lowest index, matching lax.top_k)
        int i1 = 0;
        #pragma unroll
        for (int e = 1; e < E_DIM; ++e) if (lg[e] > lg[i1]) i1 = e;
        int i2 = (i1 == 0) ? 1 : 0;
        #pragma unroll
        for (int e = 0; e < E_DIM; ++e)
            if (e != i1 && lg[e] > lg[i2]) i2 = e;

        // debias: o_j - log(K * (g_j + eps)), softmax over the 2 selected
        const float a1 = lg[i1] - __logf((float)K_SEL * (p[i1] * inv_sum + EPS_G));
        const float a2 = lg[i2] - __logf((float)K_SEL * (p[i2] * inv_sum + EPS_G));
        const float m2 = fmaxf(a1, a2);
        const float e1 = __expf(a1 - m2), e2 = __expf(a2 - m2);
        const float inv2 = 1.0f / (e1 + e2);

        float out[E_DIM];
        #pragma unroll
        for (int e = 0; e < E_DIM; ++e) out[e] = 0.f;
        out[i1] = e1 * inv2;
        out[i2] = e2 * inv2;

        f4* gp = (f4*)(gs + (size_t)row * E_DIM);
        f4 o0 = {out[0], out[1], out[2], out[3]};
        f4 o1 = {out[4], out[5], out[6], out[7]};
        gp[0] = o0;
        gp[1] = o1;
    }
}

// ---------------------------------------------------------------------------
// Kernel 2: combine  y[b,d] = sum_e h[b,d,e] * g'[b,e]
// Bandwidth-dominant: stream h (402 MB) once with NT loads (don't pollute the
// 192 MB L2), g' rows served from cache, NT store of y.
// ---------------------------------------------------------------------------
__global__ __launch_bounds__(256) void moe_combine_kernel(
    const float* __restrict__ h,    // [B, 768, 8]
    const float* __restrict__ gs,   // [B, 8]
    float* __restrict__ y,          // [B, 768]
    int total)                      // B * 768
{
    const int i = blockIdx.x * blockDim.x + threadIdx.x;
    if (i >= total) return;
    const int b = i / D_DIM;

    const f4* hp = (const f4*)(h + (size_t)i * E_DIM);
    f4 h0 = __builtin_nontemporal_load(hp);
    f4 h1 = __builtin_nontemporal_load(hp + 1);

    const f4* gp = (const f4*)(gs + (size_t)b * E_DIM);
    f4 g0 = gp[0];
    f4 g1 = gp[1];

    float acc = h0.x * g0.x + h0.y * g0.y + h0.z * g0.z + h0.w * g0.w
              + h1.x * g1.x + h1.y * g1.y + h1.z * g1.z + h1.w * g1.w;

    __builtin_nontemporal_store(acc, y + i);
}

// ---------------------------------------------------------------------------
extern "C" void kernel_launch(void* const* d_in, const int* in_sizes, int n_in,
                              void* d_out, int out_size, void* d_ws, size_t ws_size,
                              hipStream_t stream) {
    const float* h  = (const float*)d_in[0];   // [B, 768, 8]
    const float* x  = (const float*)d_in[1];   // [B, 768]
    const float* gw = (const float*)d_in[2];   // [8, 768]
    const float* bi = (const float*)d_in[3];   // [8]
    float*       y  = (float*)d_out;           // [B, 768]
    float*       gs = (float*)d_ws;            // [B, 8] sparse gates (512 KB)

    const int B = in_sizes[1] / D_DIM;         // 16384

    // Gate: one wave per 16 rows, WMMA f16->f32 GEMM + top-2 epilogue.
    moe_gate_wmma_kernel<<<B / 16, 32, 0, stream>>>(x, gw, bi, gs);

    // Combine: one thread per (b, d), streams h.
    const int total = B * D_DIM;
    moe_combine_kernel<<<(total + 255) / 256, 256, 0, stream>>>(h, gs, y, total);
}
